// HyperSchnetInteraction_23218593202760
// MI455X (gfx1250) — compile-verified, matrix operand order checked by reading, not run on hardware
//
#include <hip/hip_runtime.h>

// ---------------------------------------------------------------------------
// HyperSchnet interaction block, fused for MI455X (gfx1250, wave32, WMMA+TDM).
// B=4, A=256, F=DF=128, R=64.
// ---------------------------------------------------------------------------

typedef __attribute__((ext_vector_type(16))) _Float16 v16h;
typedef __attribute__((ext_vector_type(8)))  _Float16 v8h;
typedef __attribute__((ext_vector_type(8)))  float    v8f;
typedef __attribute__((ext_vector_type(4)))  unsigned int u32x4;
typedef __attribute__((ext_vector_type(8)))  int          i32x8;
typedef __attribute__((ext_vector_type(4)))  int          i32x4;

#define B_  4
#define A_  256
#define F_  128
#define R_  64
#define DF_ 128
#define BA_ (B_*A_)     // 1024

#if defined(__gfx1250__) && __has_builtin(__builtin_amdgcn_tensor_load_to_lds)
#define USE_TDM 1
#else
#define USE_TDM 0
#endif

// shifted softplus: log(0.5*e^v + 0.5) = max(v,0) + log(0.5 + 0.5*exp(-|v|))
__device__ __forceinline__ float ssp(float v) {
  float a = __builtin_fabsf(v);
  return __builtin_fmaxf(v, 0.0f) + __logf(0.5f + 0.5f * __expf(-a));
}

// Load a 16x32 f16 A/B WMMA fragment for this lane from a row-major [row][K]
// region. p0 must point at &base[row_of_this_lane * Kstride + kstep*32 + (lane>>4)*8].
// Per CDNA5 ISA layout: element e -> K = 32*kstep + (e>>3)*16 + (lane>>4)*8 + (e&7).
__device__ __forceinline__ v16h frag_ld(const _Float16* p0) {
  v8h lo = *(const v8h*)(p0);
  v8h hi = *(const v8h*)(p0 + 16);
  v16h r;
#pragma unroll
  for (int i = 0; i < 8; ++i) { r[i] = lo[i]; r[i + 8] = hi[i]; }
  return r;
}

__device__ __forceinline__ v8f wmma16(v16h a, v16h b, v8f c) {
  // v_wmma_f32_16x16x32_f16
  return __builtin_amdgcn_wmma_f32_16x16x32_f16(false, a, false, b,
                                                (short)0, c, false, false);
}

// ---------------------------------------------------------------------------
// Weight convert+transpose: src f32 [K][N] row-major -> dst f16 [N][K].
// ---------------------------------------------------------------------------
__global__ void k_wcvt(const float* __restrict__ src, _Float16* __restrict__ dst,
                       int K, int N) {
  int idx = blockIdx.x * blockDim.x + threadIdx.x;
  if (idx >= K * N) return;
  int k = idx / N, n = idx - k * N;
  dst[n * K + k] = (_Float16)src[idx];
}

// ---------------------------------------------------------------------------
// x_act[m][d] = ssp(x @ W_bc + b_bc), f16 output.  64 blocks x 256 threads.
// ---------------------------------------------------------------------------
__global__ void __launch_bounds__(256) k_xact(
    const float* __restrict__ x,        // [1024][128] f32
    const _Float16* __restrict__ Wbct,  // [128 n][128 k] f16
    const float* __restrict__ b_bc,     // [128]
    _Float16* __restrict__ xact)        // [1024][128] f16
{
  const int lane = threadIdx.x & 31;
  const int wave = threadIdx.x >> 5;       // tn
  const int lr   = lane & 15;
  const int hi8  = (lane >> 4) * 8;
  const int row0 = blockIdx.x * 16;

  v16h bf[4];
  const _Float16* wb = Wbct + (wave * 16 + lr) * 128;
#pragma unroll
  for (int s = 0; s < 4; ++s) bf[s] = frag_ld(wb + s * 32 + hi8);

  v8f acc = {};
#pragma unroll
  for (int s = 0; s < 4; ++s) {
    const float* xp = x + (size_t)(row0 + lr) * F_ + s * 32 + hi8;
    v16h af;
#pragma unroll
    for (int i = 0; i < 8; ++i) {
      af[i]     = (_Float16)xp[i];
      af[i + 8] = (_Float16)xp[i + 16];
    }
    acc = wmma16(af, bf[s], acc);
  }

  const int d = wave * 16 + lr;
  const float bias = b_bc[d];
#pragma unroll
  for (int r = 0; r < 8; ++r) {
    int m = row0 + r + hi8;                 // D-frag: M = r + (lane>=16 ? 8 : 0)
    xact[(size_t)m * DF_ + d] = (_Float16)ssp(acc[r] + bias);
  }
}

// ---------------------------------------------------------------------------
// Fused filter MLP + cutoff/mask weighting + neighbor-sum aggregation.
// One workgroup per (b,i): 1024 blocks x 256 threads (8 waves).
//   TDM: f_ij[b,i,:,:] (256x64 f32, 64KB)  --tensor_load_to_lds-->  sH (raw)
//   convert once:  sH(f32) -> sF(f16)
//   h1 = ssp(sF @ Wf1 + b1)  -> sH (f16, overwrites raw tile)
//   h2 = ssp(h1 @ Wf2 + b2)  -> registers only
//   y[b,i,d] = sum_j h2[j,d] * c*mask[j] * xact[b,j,d]
// 96 v_wmma per wave.  h1/h2 never hit HBM.  LDS: 97KB -> ~3 WGs/WGP.
// ---------------------------------------------------------------------------
__global__ void __launch_bounds__(256) k_filter(
    const _Float16* __restrict__ Wf1t,  // [128 n][64 k] f16
    const _Float16* __restrict__ Wf2t,  // [128 n][128 k] f16
    const float* __restrict__ b_f1,
    const float* __restrict__ b_f2,
    const float* __restrict__ f_ij,     // [B*A*A][64] f32
    const float* __restrict__ c_ij,     // [B*A*A]
    const float* __restrict__ maskp,    // [B*A*A]
    const _Float16* __restrict__ xact,  // [B*A][128] f16
    float* __restrict__ y_ws)           // [B*A][128] f32
{
  __shared__ __align__(16) _Float16 sF[A_ * R_];    // 32 KB (f16 tile)
  __shared__ __align__(16) _Float16 sH[A_ * DF_];   // 64 KB (raw f32 tile, then h1)
  __shared__ float sCM[A_];                         // 1 KB

  const int bi   = blockIdx.x;         // b*A + i
  const int b    = bi >> 8;
  const int t    = threadIdx.x;
  const int lane = t & 31;
  const int wave = t >> 5;             // tn: this wave's 16-wide column strip
  const int lr   = lane & 15;
  const int hi8  = (lane >> 4) * 8;

  // cutoff * mask
  sCM[t] = c_ij[(size_t)bi * A_ + t] * maskp[(size_t)bi * A_ + t];

  // Hint the per-batch x_act slab into cache (global_prefetch_b8).
  __builtin_prefetch(xact + (size_t)b * A_ * DF_ + t * 64, 0, 0);

#if USE_TDM
  // ---- Tensor Data Mover: DMA the contiguous 256x64 f32 slab into sH ----
  if (wave == 0) {
    unsigned long long ga =
        (unsigned long long)(uintptr_t)(f_ij + (size_t)bi * A_ * R_);
    unsigned int lds = (unsigned int)(uintptr_t)(&sH[0]);  // low 32b = LDS offset
    u32x4 g0;
    g0[0] = 1u;                                   // count=1, user mode, no gather
    g0[1] = lds;                                  // lds_addr   [63:32]
    g0[2] = (unsigned int)ga;                     // global_addr[95:64]
    g0[3] = (unsigned int)((ga >> 32) & 0x01FFFFFFu) | (2u << 30); // addr hi, type=2
    i32x8 g1;
    g1[0] = (int)0x00020000u;   // workgroup_mask=0, data_size=2 (4B), no flags
    g1[1] = (int)(64u << 16);   // tensor_dim0 = 64  (bits 79:48)
    g1[2] = (int)(256u << 16);  // tensor_dim1 = 256 (bits 111:80)
    g1[3] = (int)(64u << 16);   // tile_dim0   = 64  (bits 127:112)
    g1[4] = (int)256;           // tile_dim1   = 256 (bits 143:128), tile_dim2=0
    g1[5] = (int)64;            // tensor_dim0_stride = 64 (bits 207:160)
    g1[6] = 0;
    g1[7] = 0;                  // tensor_dim1_stride = 0 (unused, 2D tile)
    i32x4 gz  = {0, 0, 0, 0};            // groups 2/3: zero (<=2D tensor)
    i32x8 gz8 = {0, 0, 0, 0, 0, 0, 0, 0};// extra group (clang-23 6-arg form): zero
    __builtin_amdgcn_tensor_load_to_lds(g0, g1, gz, gz, gz8, 0);
    __builtin_amdgcn_s_wait_tensorcnt(0);
  }
  __syncthreads();
  // ---- one-time f32 -> f16 conversion (LDS -> LDS), coalesced b128/b64 ----
  {
    const float4* src = (const float4*)&sH[0];
#pragma unroll
    for (int it = 0; it < 16; ++it) {
      float4 v = src[it * 256 + t];
      _Float16* dp = sF + (size_t)(it * 256 + t) * 4;
      dp[0] = (_Float16)v.x; dp[1] = (_Float16)v.y;
      dp[2] = (_Float16)v.z; dp[3] = (_Float16)v.w;
    }
  }
#else
  // Fallback: stage f_ij[b,i] from global (coalesced b128) with conversion.
  {
    const float4* src = (const float4*)(f_ij + (size_t)bi * A_ * R_);
#pragma unroll
    for (int it = 0; it < 16; ++it) {
      float4 v = src[it * 256 + t];
      _Float16* dp = sF + (size_t)(it * 256 + t) * 4;
      dp[0] = (_Float16)v.x; dp[1] = (_Float16)v.y;
      dp[2] = (_Float16)v.z; dp[3] = (_Float16)v.w;
    }
  }
#endif
  __syncthreads();   // sF ready; sH free to be overwritten by h1

  // Preload this wave's weight B-fragments (held in VGPRs for all 16 row tiles).
  v16h w1[2], w2[4];
  {
    const _Float16* p1 = Wf1t + (wave * 16 + lr) * R_;
    w1[0] = frag_ld(p1 + hi8);
    w1[1] = frag_ld(p1 + 32 + hi8);
    const _Float16* p2 = Wf2t + (wave * 16 + lr) * DF_;
#pragma unroll
    for (int s = 0; s < 4; ++s) w2[s] = frag_ld(p2 + s * 32 + hi8);
  }
  const float bias1 = b_f1[wave * 16 + lr];
  const float bias2 = b_f2[wave * 16 + lr];
  const int   d     = wave * 16 + lr;

  // ---- GEMM1: 16 row tiles x 2 k-steps -> ssp -> sH (f16) ----
#pragma unroll 4
  for (int tm = 0; tm < 16; ++tm) {
    const _Float16* ap = sF + (size_t)(tm * 16 + lr) * R_;
    v8f acc = {};
    acc = wmma16(frag_ld(ap + hi8),      w1[0], acc);
    acc = wmma16(frag_ld(ap + 32 + hi8), w1[1], acc);
    _Float16* hp = sH + (size_t)(tm * 16 + hi8) * DF_ + d;
#pragma unroll
    for (int r = 0; r < 8; ++r)
      hp[(size_t)r * DF_] = (_Float16)ssp(acc[r] + bias1);
  }
  __syncthreads();

  // ---- GEMM2 + weighted aggregation over j (register-resident h2) ----
  const _Float16* xa = xact + (size_t)b * A_ * DF_ + d;
  float ysum = 0.0f;
#pragma unroll 2
  for (int tm = 0; tm < 16; ++tm) {
    const _Float16* ap = sH + (size_t)(tm * 16 + lr) * DF_;
    v8f acc = {};
#pragma unroll
    for (int s = 0; s < 4; ++s)
      acc = wmma16(frag_ld(ap + s * 32 + hi8), w2[s], acc);
#pragma unroll
    for (int r = 0; r < 8; ++r) {
      int j = tm * 16 + hi8 + r;           // D-frag row
      float h2 = ssp(acc[r] + bias2);
      ysum += h2 * sCM[j] * (float)xa[(size_t)j * DF_];
    }
  }
  // Lanes L and L^16 hold the same column d -> combine halves.
  ysum += __shfl_xor(ysum, 16, 32);
  if (lane < 16) y_ws[(size_t)bi * DF_ + d] = ysum;
}

// ---------------------------------------------------------------------------
// Final atomwise MLP + residual: out = x + ssp(ssp(y@Wa1+b1)@Wa2+b2).
// ---------------------------------------------------------------------------
__global__ void __launch_bounds__(256) k_out(
    const float* __restrict__ y_ws,      // [1024][128] f32
    const _Float16* __restrict__ Wa1t,   // [128][128] f16
    const _Float16* __restrict__ Wa2t,   // [128][128] f16
    const float* __restrict__ b_a1,
    const float* __restrict__ b_a2,
    const float* __restrict__ x,         // [1024][128] f32
    float* __restrict__ out)             // [1024][128] f32
{
  __shared__ __align__(16) _Float16 sT[16 * F_];

  const int lane = threadIdx.x & 31;
  const int wave = threadIdx.x >> 5;
  const int lr   = lane & 15;
  const int hi8  = (lane >> 4) * 8;
  const int row0 = blockIdx.x * 16;
  const int d    = wave * 16 + lr;

  v16h wa1[4], wa2[4];
  {
    const _Float16* p1 = Wa1t + (size_t)d * F_;
    const _Float16* p2 = Wa2t + (size_t)d * F_;
#pragma unroll
    for (int s = 0; s < 4; ++s) {
      wa1[s] = frag_ld(p1 + s * 32 + hi8);
      wa2[s] = frag_ld(p2 + s * 32 + hi8);
    }
  }
  const float bias1 = b_a1[d];
  const float bias2 = b_a2[d];

  // layer 1: A from y_ws (f32 -> f16 on the fly)
  v8f acc = {};
#pragma unroll
  for (int s = 0; s < 4; ++s) {
    const float* yp = y_ws + (size_t)(row0 + lr) * F_ + s * 32 + hi8;
    v16h af;
#pragma unroll
    for (int i = 0; i < 8; ++i) {
      af[i]     = (_Float16)yp[i];
      af[i + 8] = (_Float16)yp[i + 16];
    }
    acc = wmma16(af, wa1[s], acc);
  }
  {
    _Float16* hp = sT + (size_t)hi8 * F_ + d;
#pragma unroll
    for (int r = 0; r < 8; ++r)
      hp[(size_t)r * F_] = (_Float16)ssp(acc[r] + bias1);
  }
  __syncthreads();

  // layer 2 + residual
  v8f acc2 = {};
  const _Float16* ap = sT + (size_t)lr * F_;
#pragma unroll
  for (int s = 0; s < 4; ++s)
    acc2 = wmma16(frag_ld(ap + s * 32 + hi8), wa2[s], acc2);
#pragma unroll
  for (int r = 0; r < 8; ++r) {
    int m = row0 + r + hi8;
    out[(size_t)m * F_ + d] = x[(size_t)m * F_ + d] + ssp(acc2[r] + bias2);
  }
}

// ---------------------------------------------------------------------------
// Host-side launch.  Inputs (setup_inputs order):
//  0:x 1:f_ij 2:c_ij 3:mask 4:W_bc 5:b_bc 6:W_f1 7:b_f1 8:W_f2 9:b_f2
// 10:W_a1 11:b_a1 12:W_a2 13:b_a2
// ---------------------------------------------------------------------------
extern "C" void kernel_launch(void* const* d_in, const int* in_sizes, int n_in,
                              void* d_out, int out_size, void* d_ws, size_t ws_size,
                              hipStream_t stream) {
  const float* x    = (const float*)d_in[0];
  const float* f_ij = (const float*)d_in[1];
  const float* c_ij = (const float*)d_in[2];
  const float* mask = (const float*)d_in[3];
  const float* W_bc = (const float*)d_in[4];
  const float* b_bc = (const float*)d_in[5];
  const float* W_f1 = (const float*)d_in[6];
  const float* b_f1 = (const float*)d_in[7];
  const float* W_f2 = (const float*)d_in[8];
  const float* b_f2 = (const float*)d_in[9];
  const float* W_a1 = (const float*)d_in[10];
  const float* b_a1 = (const float*)d_in[11];
  const float* W_a2 = (const float*)d_in[12];
  const float* b_a2 = (const float*)d_in[13];
  float* out = (float*)d_out;

  // Workspace layout (bytes):
  char* ws = (char*)d_ws;
  _Float16* Wbct = (_Float16*)(ws + 0);         // 128*128*2 = 32768
  _Float16* Wf1t = (_Float16*)(ws + 32768);     // 128*64*2  = 16384
  _Float16* Wf2t = (_Float16*)(ws + 49152);     // 32768
  _Float16* Wa1t = (_Float16*)(ws + 81920);     // 32768
  _Float16* Wa2t = (_Float16*)(ws + 114688);    // 32768
  _Float16* xact = (_Float16*)(ws + 147456);    // 1024*128*2 = 262144
  float*    y_ws = (float*)   (ws + 409600);    // 1024*128*4 = 524288
  // total: 933888 bytes

  // 1) weight f32->f16 transposes
  k_wcvt<<<(F_*F_ + 255) / 256, 256, 0, stream>>>(W_bc, Wbct, F_, DF_);
  k_wcvt<<<(R_*DF_ + 255) / 256, 256, 0, stream>>>(W_f1, Wf1t, R_, DF_);
  k_wcvt<<<(DF_*DF_ + 255) / 256, 256, 0, stream>>>(W_f2, Wf2t, DF_, DF_);
  k_wcvt<<<(DF_*F_ + 255) / 256, 256, 0, stream>>>(W_a1, Wa1t, DF_, F_);
  k_wcvt<<<(F_*F_ + 255) / 256, 256, 0, stream>>>(W_a2, Wa2t, F_, F_);

  // 2) x_act = ssp(x @ W_bc + b_bc)  (f16)
  k_xact<<<BA_ / 16, 256, 0, stream>>>(x, Wbct, b_bc, xact);

  // 3) fused filter MLP + aggregation (one workgroup per (b,i))
  k_filter<<<BA_, 256, 0, stream>>>(Wf1t, Wf2t, b_f1, b_f2,
                                    f_ij, c_ij, mask, xact, y_ws);

  // 4) final MLP + residual
  k_out<<<BA_ / 16, 256, 0, stream>>>(y_ws, Wa1t, Wa2t, b_a1, b_a2, x, out);
}
